// GINEEncoder_39822936768758
// MI455X (gfx1250) — matrix operand order repaired and checked by reading, not compiled, not run on hardware
//
#include <hip/hip_runtime.h>
#include <hip/hip_bf16.h>
#include <math.h>

// ---------------------------------------------------------------------------
// Types for CDNA5 WMMA (wave32): v_wmma_f32_16x16x32_bf16
// ---------------------------------------------------------------------------
typedef __bf16 bf16_t;
typedef __attribute__((ext_vector_type(16))) __bf16 v16bf;
typedef __attribute__((ext_vector_type(4)))  __bf16 v4bf;
typedef __attribute__((ext_vector_type(8)))  float  v8f;

#define LN_EPS 1e-5f

__device__ __forceinline__ float gelu_exact(float x) {
    return 0.5f * x * (1.0f + erff(x * 0.70710678118654752f));
}
__device__ __forceinline__ float sigmoidf_(float x) {
    return 1.0f / (1.0f + __expf(-x));
}

// ---------------------------------------------------------------------------
// Weight pre-swizzle: f32 row-major W[K][N] -> bf16 per-lane B-fragment order.
// Layout: Wb[(((kt*numNt + nt)*32 + lane)*16 + e)]
//   col    = nt*16 + (lane & 15)
//   klocal = (lane>>4)*16 + e           (B fragment: 16 consecutive K per lane)
// Zero-pads K up to a multiple of 32.
// ---------------------------------------------------------------------------
__global__ void convert_w_kernel(const float* __restrict__ W, bf16_t* __restrict__ Wb,
                                 int K, int N) {
    int numNt = N >> 4;
    int numKt = (K + 31) >> 5;
    int total = numKt * numNt * 512;
    int i = blockIdx.x * blockDim.x + threadIdx.x;
    if (i >= total) return;
    int e  = i & 15;
    int l  = (i >> 4) & 31;
    int nt = (i >> 9) % numNt;
    int kt = (i >> 9) / numNt;
    int col = nt * 16 + (l & 15);
    int k   = kt * 32 + ((l >> 4) << 4) + e;
    Wb[i] = (bf16_t)((k < K) ? W[(size_t)k * N + col] : 0.0f);
}

// ---------------------------------------------------------------------------
// Per-row LayerNorm stats (mean, rstd): one wave per row.
// ---------------------------------------------------------------------------
__global__ void row_stats_kernel(const float* __restrict__ A, int M, int D,
                                 float* __restrict__ mean, float* __restrict__ rstd) {
    int row  = blockIdx.x * (blockDim.x >> 5) + (threadIdx.x >> 5);
    int lane = threadIdx.x & 31;
    if (row >= M) return;
    const float* a = A + (size_t)row * D;
    float s = 0.0f, s2 = 0.0f;
    for (int k = lane; k < D; k += 32) { float v = a[k]; s += v; s2 += v * v; }
    for (int o = 16; o > 0; o >>= 1) { s += __shfl_xor(s, o, 32); s2 += __shfl_xor(s2, o, 32); }
    if (lane == 0) {
        float m = s / (float)D;
        float var = s2 / (float)D - m * m;
        mean[row] = m;
        rstd[row] = rsqrtf(var + LN_EPS);
    }
}

// ---------------------------------------------------------------------------
// Generic WMMA GEMM:  C[:, col_off:col_off+N] = epi( LN?(A) @ W + bias )
//   block = 128 threads = 4 waves; wave w computes n-tiles (blockIdx.y*16+4w ..+3)
//   A fragment (16 rows x 32 K, bf16) built cooperatively in LDS with fused
//   LayerNorm; all 4 waves share it (ds_load_b128).
//
//   WMMA section is branch-free: out-of-range n-tiles are clamped to the last
//   valid tile (EXEC stays all-1s as the ISA requires); range checks happen
//   only at the store.  epi: 0 = none, 1 = GELU(erf), 2 = sigmoid
//
//   Fill: each thread owns 4 CONSECUTIVE elements of ONE lane's fragment:
//   same row, consecutive K -> one float4 load + one ds_store_b64 fast path,
//   with LN row constants hoisted out of the K loop.
// ---------------------------------------------------------------------------
__global__ void __launch_bounds__(128)
gemm_wmma(const float* __restrict__ A, int lda, int M, int K,
          const bf16_t* __restrict__ Wb, int N, const float* __restrict__ bias,
          const float* __restrict__ mean, const float* __restrict__ rstd,
          const float* __restrict__ lng, const float* __restrict__ lnb,
          float* __restrict__ C, int ldc, int col_off, int epi) {
    __shared__ __attribute__((aligned(32))) bf16_t afrag[512];

    const int mt   = blockIdx.x;
    const int wave = threadIdx.x >> 5;
    const int lane = threadIdx.x & 31;
    const int numNt = N >> 4;
    const int numKt = (K + 31) >> 5;
    const int ntBase = blockIdx.y * 16 + wave * 4;

    // Clamp tile indices so WMMA + its loads run unconditionally (EXEC all-1s).
    int tileOff[4];
    #pragma unroll
    for (int t = 0; t < 4; ++t) {
        int nt = ntBase + t;
        int ntc = (nt < numNt) ? nt : (numNt - 1);
        tileOff[t] = ntc * 512;                 // elements into the kt-slice
    }

    // --- fill-role constants (per thread, hoisted out of the K loop) ---
    const int tgrp   = threadIdx.x << 2;        // 0,4,...,508: 4 frag elems
    const int flane  = tgrp >> 4;               // target lane of these elems
    const int e0     = tgrp & 15;               // first element index (0/4/8/12)
    const int frow   = mt * 16 + (flane & 15);  // source row (constant!)
    // A-frag K mapping: contiguous k = 16*(lane>>4) + e (matches B); 4 elems are consecutive K:
    const int klbase = ((flane >> 4) << 4) + e0;
    const bool rowOK = (frow < M);
    float mn = 0.0f, rs = 1.0f;
    if (mean && rowOK) { mn = mean[frow]; rs = rstd[frow]; }
    const float* arow = A + (size_t)frow * lda;

    v8f acc0 = {}, acc1 = {}, acc2 = {}, acc3 = {};

    for (int kt = 0; kt < numKt; ++kt) {
        // ---- cooperative A-fragment fill (LN fused, f32 -> bf16) ----
        int k0 = kt * 32 + klbase;
        float4 va = make_float4(0.f, 0.f, 0.f, 0.f);
        if (rowOK) {
            if (k0 + 3 < K) {                   // aligned fast path
                va = *(const float4*)(arow + k0);
            } else {                            // ragged tail (K=12/20/50 pads)
                if (k0 + 0 < K) va.x = arow[k0 + 0];
                if (k0 + 1 < K) va.y = arow[k0 + 1];
                if (k0 + 2 < K) va.z = arow[k0 + 2];
                if (k0 + 3 < K) va.w = arow[k0 + 3];
            }
        }
        if (mean) {
            va.x = (va.x - mn) * rs; va.y = (va.y - mn) * rs;
            va.z = (va.z - mn) * rs; va.w = (va.w - mn) * rs;
        }
        if (lng) {
            float4 g4 = make_float4(1.f, 1.f, 1.f, 1.f);
            float4 b4 = make_float4(0.f, 0.f, 0.f, 0.f);
            if (k0 + 3 < K) {
                g4 = *(const float4*)(lng + k0);
                b4 = *(const float4*)(lnb + k0);
            } else {
                if (k0 + 0 < K) { g4.x = lng[k0 + 0]; b4.x = lnb[k0 + 0]; }
                if (k0 + 1 < K) { g4.y = lng[k0 + 1]; b4.y = lnb[k0 + 1]; }
                if (k0 + 2 < K) { g4.z = lng[k0 + 2]; b4.z = lnb[k0 + 2]; }
                if (k0 + 3 < K) { g4.w = lng[k0 + 3]; b4.w = lnb[k0 + 3]; }
            }
            va.x = va.x * g4.x + b4.x; va.y = va.y * g4.y + b4.y;
            va.z = va.z * g4.z + b4.z; va.w = va.w * g4.w + b4.w;
        }
        v4bf packed;
        packed[0] = (bf16_t)va.x; packed[1] = (bf16_t)va.y;
        packed[2] = (bf16_t)va.z; packed[3] = (bf16_t)va.w;
        *(v4bf*)(afrag + tgrp) = packed;        // one ds_store_b64
        __syncthreads();

        v16bf a = *(const v16bf*)(afrag + lane * 16);
        const bf16_t* kbase = Wb + (size_t)kt * numNt * 512 + lane * 16;

        v16bf b0 = *(const v16bf*)(kbase + tileOff[0]);
        v16bf b1 = *(const v16bf*)(kbase + tileOff[1]);
        v16bf b2 = *(const v16bf*)(kbase + tileOff[2]);
        v16bf b3 = *(const v16bf*)(kbase + tileOff[3]);
        acc0 = __builtin_amdgcn_wmma_f32_16x16x32_bf16(false, a, false, b0, (short)0, acc0, false, false);
        acc1 = __builtin_amdgcn_wmma_f32_16x16x32_bf16(false, a, false, b1, (short)0, acc1, false, false);
        acc2 = __builtin_amdgcn_wmma_f32_16x16x32_bf16(false, a, false, b2, (short)0, acc2, false, false);
        acc3 = __builtin_amdgcn_wmma_f32_16x16x32_bf16(false, a, false, b3, (short)0, acc3, false, false);
        __syncthreads();
    }

    // ---- epilogue: bias + activation + store (range checks safe here) ----
    const int rbase = mt * 16 + ((lane >> 4) << 3);
    v8f accs[4] = {acc0, acc1, acc2, acc3};
    #pragma unroll
    for (int t = 0; t < 4; ++t) {
        int nt = ntBase + t;
        if (nt < numNt) {
            int col = nt * 16 + (lane & 15);
            float bv = bias ? bias[col] : 0.0f;
            #pragma unroll
            for (int r = 0; r < 8; ++r) {
                int row = rbase + r;
                if (row < M) {
                    float v = accs[t][r] + bv;
                    if (epi == 1)      v = gelu_exact(v);
                    else if (epi == 2) v = sigmoidf_(v);
                    C[(size_t)row * ldc + col_off + col] = v;
                }
            }
        }
    }
}

// ---------------------------------------------------------------------------
// Edge raw features: [RBF(32) | seq_emb(16) | is_seq | inv_dist | pad to 64]
// ---------------------------------------------------------------------------
__global__ void edge_raw_kernel(const float* __restrict__ dist,
                                const float* __restrict__ is_seq,
                                const float* __restrict__ inv_dist,
                                const int* __restrict__ seqbin,
                                const float* __restrict__ seq_emb,
                                float* __restrict__ ER, int E) {
    int i = blockIdx.x * blockDim.x + threadIdx.x;
    if (i >= E * 64) return;
    int e = i >> 6, c = i & 63;
    float v = 0.0f;
    if (c < 32) {
        float center = 20.0f * (float)c / 31.0f;               // linspace(0,20,32)
        float step   = 20.0f / 32.0f;
        float gamma  = 1.0f / (step * step + 1e-8f);
        float d = dist[e] - center;
        v = __expf(-gamma * d * d);
    } else if (c < 48) {
        v = seq_emb[seqbin[e] * 16 + (c - 32)];
    } else if (c == 48) {
        v = is_seq[e];
    } else if (c == 49) {
        v = inv_dist[e];
    }
    ER[(size_t)e * 64 + c] = v;
}

// ---------------------------------------------------------------------------
// Fuse-input build: F = [ G[:,0:128] | G[:,128:384] * GATE ]
// ---------------------------------------------------------------------------
__global__ void fbuild_kernel(const float* __restrict__ G, const float* __restrict__ GATE,
                              float* __restrict__ F, int Nn) {
    int i = blockIdx.x * blockDim.x + threadIdx.x;
    if (i >= Nn * 384) return;
    int r = i / 384, c = i - r * 384;
    float v = G[i];
    if (c >= 128) v *= GATE[(size_t)r * 256 + (c - 128)];
    F[i] = v;
}

// ---------------------------------------------------------------------------
// GINE message pass: AGG[tgt] += relu(X[src] + EA[e])   (AGG pre-seeded with X)
// One thread per (edge, 4-column chunk).
// ---------------------------------------------------------------------------
__global__ void scatter_kernel(const float* __restrict__ X, const float* __restrict__ EA,
                               const int* __restrict__ eidx, float* __restrict__ AGG, int E) {
    int i = blockIdx.x * blockDim.x + threadIdx.x;
    if (i >= E * 64) return;
    int e = i >> 6, c = (i & 63) << 2;
    int s = eidx[e], t = eidx[E + e];
    const float4 xv = *(const float4*)(X + (size_t)s * 256 + c);
    const float4 ev = *(const float4*)(EA + (size_t)e * 256 + c);
    float m0 = fmaxf(xv.x + ev.x, 0.0f);
    float m1 = fmaxf(xv.y + ev.y, 0.0f);
    float m2 = fmaxf(xv.z + ev.z, 0.0f);
    float m3 = fmaxf(xv.w + ev.w, 0.0f);
    float* dst = AGG + (size_t)t * 256 + c;
    atomicAdd(dst + 0, m0);
    atomicAdd(dst + 1, m1);
    atomicAdd(dst + 2, m2);
    atomicAdd(dst + 3, m3);
}

// ---------------------------------------------------------------------------
// Post-conv: Xout = gelu(LN(H)*g+b) + Xin   (one wave per 256-wide row)
// ---------------------------------------------------------------------------
__global__ void postnorm_kernel(const float* __restrict__ H, const float* __restrict__ Xin,
                                const float* __restrict__ g, const float* __restrict__ b,
                                float* __restrict__ Xout, int Nn) {
    int row  = blockIdx.x * (blockDim.x >> 5) + (threadIdx.x >> 5);
    int lane = threadIdx.x & 31;
    if (row >= Nn) return;
    const float* h = H + (size_t)row * 256;
    float v[8], s = 0.0f, s2 = 0.0f;
    #pragma unroll
    for (int j = 0; j < 8; ++j) { float x = h[lane + j * 32]; v[j] = x; s += x; s2 += x * x; }
    for (int o = 16; o > 0; o >>= 1) { s += __shfl_xor(s, o, 32); s2 += __shfl_xor(s2, o, 32); }
    float m = s * (1.0f / 256.0f);
    float rs = rsqrtf(s2 * (1.0f / 256.0f) - m * m + LN_EPS);
    #pragma unroll
    for (int j = 0; j < 8; ++j) {
        int c = lane + j * 32;
        float y = (v[j] - m) * rs * g[c] + b[c];
        Xout[(size_t)row * 256 + c] = gelu_exact(y) + Xin[(size_t)row * 256 + c];
    }
}

// ---------------------------------------------------------------------------
// Readout: per-graph sum / max / count, then hcat = [mean | max]  [32, 512]
// ---------------------------------------------------------------------------
__device__ __forceinline__ void atomicMaxF(float* addr, float v) {
    if (v >= 0.0f) atomicMax((int*)addr, __float_as_int(v));
    else           atomicMin((unsigned int*)addr, (unsigned int)__float_as_int(v));
}

__global__ void readout_init_kernel(float* sums, float* maxs, float* counts) {
    int i = blockIdx.x * blockDim.x + threadIdx.x;
    if (i < 32 * 256) { sums[i] = 0.0f; maxs[i] = -INFINITY; }
    if (i < 32) counts[i] = 0.0f;
}

__global__ void readout_accum_kernel(const float* __restrict__ X, const int* __restrict__ batch,
                                     float* __restrict__ sums, float* __restrict__ maxs,
                                     float* __restrict__ counts, int Nn) {
    int i = blockIdx.x * blockDim.x + threadIdx.x;
    if (i >= Nn * 64) return;
    int n = i >> 6, c = (i & 63) << 2;
    int g = batch[n];
    const float4 xv = *(const float4*)(X + (size_t)n * 256 + c);
    float* sp = sums + (size_t)g * 256 + c;
    float* mp = maxs + (size_t)g * 256 + c;
    atomicAdd(sp + 0, xv.x); atomicAdd(sp + 1, xv.y);
    atomicAdd(sp + 2, xv.z); atomicAdd(sp + 3, xv.w);
    atomicMaxF(mp + 0, xv.x); atomicMaxF(mp + 1, xv.y);
    atomicMaxF(mp + 2, xv.z); atomicMaxF(mp + 3, xv.w);
    if ((i & 63) == 0) atomicAdd(&counts[g], 1.0f);
}

__global__ void hcat_kernel(const float* __restrict__ sums, const float* __restrict__ maxs,
                            const float* __restrict__ counts, float* __restrict__ hcat) {
    int i = blockIdx.x * blockDim.x + threadIdx.x;
    if (i >= 32 * 256) return;
    int g = i >> 8, c = i & 255;
    float cnt = counts[g];
    hcat[(size_t)g * 512 + c]       = sums[i] / fmaxf(cnt, 1.0f);
    hcat[(size_t)g * 512 + 256 + c] = (cnt > 0.0f) ? maxs[i] : 0.0f;
}

// ---------------------------------------------------------------------------
// Host orchestration
// ---------------------------------------------------------------------------
extern "C" void kernel_launch(void* const* d_in, const int* in_sizes, int n_in,
                              void* d_out, int out_size, void* d_ws, size_t ws_size,
                              hipStream_t stream) {
    (void)in_sizes; (void)n_in; (void)out_size; (void)ws_size;
    const int N = 50000, E = 200000;

    const float* x_base  = (const float*)d_in[0];
    const float* x_b62   = (const float*)d_in[1];
    const float* x_esm   = (const float*)d_in[2];
    const float* e_dist  = (const float*)d_in[3];
    const float* e_isseq = (const float*)d_in[4];
    const float* e_invd  = (const float*)d_in[5];
    const int*   e_idx   = (const int*)d_in[6];
    const int*   e_bin   = (const int*)d_in[7];
    const int*   batch   = (const int*)d_in[8];
    #define PP(i) ((const float*)d_in[9 + (i)])
    // param leaf order (insertion order of nested dicts):
    // base: 0 g,1 b,2 w1,3 b1,4 w2,5 b2 | b62: 6..11 | esm: 12..17
    // struct: 18 g,19 b,20 w,21 bb | gate: 22 w1,23 b1,24 w2,25 b2
    // fuse: 26 g,27 b,28 w,29 bb | 30 seq_emb | edge: 31 w1,32 b1,33 w2,34 b2
    // convs[i]: 35+4i (w1,b1,w2,b2) | norms[i]: 55+2i (g,b) | readout: 65 w,66 bb

    char* ws = (char*)d_ws;
    size_t off = 0;
    auto alloc = [&](size_t bytes) -> void* {
        void* p = ws + off;
        off = (off + bytes + 255) & ~(size_t)255;
        return p;
    };

    auto convw = [&](const float* W, int K, int Nn) -> bf16_t* {
        int numKt = (K + 31) / 32, numNt = Nn / 16;
        int total = numKt * numNt * 512;
        bf16_t* Wb = (bf16_t*)alloc((size_t)total * sizeof(bf16_t));
        convert_w_kernel<<<(total + 255) / 256, 256, 0, stream>>>(W, Wb, K, Nn);
        return Wb;
    };
    auto gemm = [&](const float* A, int lda, int M, int K, const bf16_t* Wb, int Nn,
                    const float* bias, const float* mean, const float* rstd,
                    const float* lng, const float* lnb,
                    float* C, int ldc, int coloff, int epi) {
        dim3 grid((M + 15) / 16, (Nn + 255) / 256);
        gemm_wmma<<<grid, 128, 0, stream>>>(A, lda, M, K, Wb, Nn, bias,
                                            mean, rstd, lng, lnb, C, ldc, coloff, epi);
    };
    auto stats = [&](const float* A, int M, int D, float* mean, float* rstd) {
        row_stats_kernel<<<(M + 3) / 4, 128, 0, stream>>>(A, M, D, mean, rstd);
    };

    // ---- weight swizzle (bf16 fragment layout) ----
    bf16_t* base_w1b = convw(PP(2), 12, 64);
    bf16_t* base_w2b = convw(PP(4), 64, 64);
    bf16_t* b62_w1b  = convw(PP(8), 20, 64);
    bf16_t* b62_w2b  = convw(PP(10), 64, 64);
    bf16_t* esm_w1b  = convw(PP(14), 1280, 256);
    bf16_t* esm_w2b  = convw(PP(16), 256, 256);
    bf16_t* st_wb    = convw(PP(20), 128, 128);
    bf16_t* gate_w1b = convw(PP(22), 384, 256);
    bf16_t* gate_w2b = convw(PP(24), 256, 256);
    bf16_t* fuse_wb  = convw(PP(28), 384, 256);
    bf16_t* edge_w1b = convw(PP(31), 50, 256);
    bf16_t* edge_w2b = convw(PP(33), 256, 256);
    bf16_t* conv_w1b[5], *conv_w2b[5];
    for (int i = 0; i < 5; ++i) {
        conv_w1b[i] = convw(PP(35 + 4 * i), 256, 256);
        conv_w2b[i] = convw(PP(37 + 4 * i), 256, 256);
    }
    bf16_t* ro_wb = convw(PP(65), 512, 256);

    // ---- activation buffers ----
    float* mean  = (float*)alloc((size_t)N * 4);
    float* rstd  = (float*)alloc((size_t)N * 4);
    float* T64   = (float*)alloc((size_t)N * 64 * 4);
    float* SIn   = (float*)alloc((size_t)N * 128 * 4);
    float* G     = (float*)alloc((size_t)N * 384 * 4);
    float* T1    = (float*)alloc((size_t)N * 256 * 4);
    float* GATE  = (float*)alloc((size_t)N * 256 * 4);
    float* F     = (float*)alloc((size_t)N * 384 * 4);
    float* X     = (float*)alloc((size_t)N * 256 * 4);
    float* ER    = (float*)alloc((size_t)E * 64 * 4);
    float* ET    = (float*)alloc((size_t)E * 256 * 4);   // reused as H/AGG/X2 later
    float* EA    = (float*)alloc((size_t)E * 256 * 4);
    float* sums  = (float*)alloc(32 * 256 * 4);
    float* maxs  = (float*)alloc(32 * 256 * 4);
    float* cnts  = (float*)alloc(32 * 4);
    float* hcat  = (float*)alloc(32 * 512 * 4);

    // ---- node feature pipeline ----
    // base MLP2: LN -> w1 -> gelu -> w2 -> gelu  -> SIn[:,0:64]
    stats(x_base, N, 12, mean, rstd);
    gemm(x_base, 12, N, 12, base_w1b, 64, PP(3), mean, rstd, PP(0), PP(1), T64, 64, 0, 1);
    gemm(T64, 64, N, 64, base_w2b, 64, PP(5), nullptr, nullptr, nullptr, nullptr, SIn, 128, 0, 1);
    // b62 MLP2 -> SIn[:,64:128]
    stats(x_b62, N, 20, mean, rstd);
    gemm(x_b62, 20, N, 20, b62_w1b, 64, PP(9), mean, rstd, PP(6), PP(7), T64, 64, 0, 1);
    gemm(T64, 64, N, 64, b62_w2b, 64, PP(11), nullptr, nullptr, nullptr, nullptr, SIn, 128, 64, 1);
    // struct: gelu(LN(SIn) @ w + bb) -> G[:,0:128]
    stats(SIn, N, 128, mean, rstd);
    gemm(SIn, 128, N, 128, st_wb, 128, PP(21), mean, rstd, PP(18), PP(19), G, 384, 0, 1);
    // esm MLP2 -> G[:,128:384]
    stats(x_esm, N, 1280, mean, rstd);
    gemm(x_esm, 1280, N, 1280, esm_w1b, 256, PP(15), mean, rstd, PP(12), PP(13), T1, 256, 0, 1);
    gemm(T1, 256, N, 256, esm_w2b, 256, PP(17), nullptr, nullptr, nullptr, nullptr, G, 384, 128, 1);
    // gate = sigmoid(gelu(G @ gw1 + gb1) @ gw2 + gb2)
    gemm(G, 384, N, 384, gate_w1b, 256, PP(23), nullptr, nullptr, nullptr, nullptr, T1, 256, 0, 1);
    gemm(T1, 256, N, 256, gate_w2b, 256, PP(25), nullptr, nullptr, nullptr, nullptr, GATE, 256, 0, 2);
    // F = [h_struct | h_esm * gate]; x = gelu(LN(F) @ fuse_w + bb)
    fbuild_kernel<<<((size_t)N * 384 + 255) / 256, 256, 0, stream>>>(G, GATE, F, N);
    stats(F, N, 384, mean, rstd);
    gemm(F, 384, N, 384, fuse_wb, 256, PP(29), mean, rstd, PP(26), PP(27), X, 256, 0, 1);

    // ---- edge pipeline ----
    edge_raw_kernel<<<((size_t)E * 64 + 255) / 256, 256, 0, stream>>>(
        e_dist, e_isseq, e_invd, e_bin, PP(30), ER, E);
    gemm(ER, 64, E, 50, edge_w1b, 256, PP(32), nullptr, nullptr, nullptr, nullptr, ET, 256, 0, 1);
    gemm(ET, 256, E, 256, edge_w2b, 256, PP(34), nullptr, nullptr, nullptr, nullptr, EA, 256, 0, 0);

    // ---- conv layers (reuse ET region: dead after EA computed) ----
    float* H   = ET;
    float* AGG = ET + (size_t)N * 256;
    float* X2  = ET + (size_t)2 * N * 256;
    for (int i = 0; i < 5; ++i) {
        hipMemcpyAsync(AGG, X, (size_t)N * 256 * 4, hipMemcpyDeviceToDevice, stream);
        scatter_kernel<<<((size_t)E * 64 + 255) / 256, 256, 0, stream>>>(X, EA, e_idx, AGG, E);
        gemm(AGG, 256, N, 256, conv_w1b[i], 256, PP(36 + 4 * i),
             nullptr, nullptr, nullptr, nullptr, T1, 256, 0, 1);
        gemm(T1, 256, N, 256, conv_w2b[i], 256, PP(38 + 4 * i),
             nullptr, nullptr, nullptr, nullptr, H, 256, 0, 0);
        postnorm_kernel<<<(N + 3) / 4, 128, 0, stream>>>(H, X, PP(55 + 2 * i), PP(56 + 2 * i), X2, N);
        float* tmp = X; X = X2; X2 = tmp;
    }

    // ---- readout ----
    readout_init_kernel<<<(32 * 256 + 255) / 256, 256, 0, stream>>>(sums, maxs, cnts);
    readout_accum_kernel<<<((size_t)N * 64 + 255) / 256, 256, 0, stream>>>(X, batch, sums, maxs, cnts, N);
    hcat_kernel<<<(32 * 256 + 255) / 256, 256, 0, stream>>>(sums, maxs, cnts, hcat);
    gemm(hcat, 512, 32, 512, ro_wb, 256, PP(66), nullptr, nullptr, nullptr, nullptr,
         (float*)d_out, 256, 0, 1);
    #undef PP
}
